// ImageEncoderViT_24919400251626
// MI455X (gfx1250) — compile-verified, matrix operand order checked by reading, not run on hardware
//
#include <hip/hip_runtime.h>
#include <hip/hip_bf16.h>
#include <math.h>

typedef __attribute__((ext_vector_type(16))) _Float16 v16h;
typedef __attribute__((ext_vector_type(8)))  float    v8f;

// CDNA5 16-bit A/B operand K-index map (ISA 7.12.2):
// lane half-group g (lane/16), element e of v16h (e=2v+p):
//   v<4 : K = 2v + 8g + p        (K 0..15 region)
//   v>=4: K = 16 + 2(v-4) + 8g+p (K 16..31 region)
__device__ __forceinline__ int kmap(int e, int g) {
    int v = e >> 1, p = e & 1;
    return ((v < 4) ? (2 * v + 8 * g) : (16 + 2 * (v - 4) + 8 * g)) + p;
}

#define WMMA_F16(a, b, c) __builtin_amdgcn_wmma_f32_16x16x32_f16( \
    false, (a), false, (b), (short)0, (c), false, false)

// ---------------------------------------------------------------------------
// Generic tiled GEMM: C[M,N] = A[M,K] * B[K,N]  (A,B fp32 in HBM -> f16 LDS)
// 128 threads = 4 waves; block tile 128x64; k-step 32; wave tile 32x64 via
// 8x v_wmma_f32_16x16x32_f16 per k-step. global_prefetch of next k-tile.
// Epilogue flags: bias / GELU / residual.
// ---------------------------------------------------------------------------
#define EP_BIAS 1
#define EP_GELU 2
#define EP_RES  4

template <int EP>
__global__ void __launch_bounds__(128)
gemm_f16wmma(const float* __restrict__ A, const float* __restrict__ B,
             const float* __restrict__ bias, const float* __restrict__ res,
             float* __restrict__ C, int M, int N, int K)
{
    __shared__ _Float16 As[128][40];  // [row][k]
    __shared__ _Float16 Bs[64][40];   // transposed: [n][k]
    const int tid  = threadIdx.x;
    const int wave = tid >> 5;
    const int lane = tid & 31;
    const int g    = lane >> 4;
    const int ln   = lane & 15;
    const int bm   = blockIdx.y * 128;
    const int bn   = blockIdx.x * 64;

    v8f acc[2][4] = {};

    for (int k0 = 0; k0 < K; k0 += 32) {
        // stage A tile 128x32
        #pragma unroll 4
        for (int i = 0; i < 32; ++i) {
            int flat = tid + i * 128;
            int r = flat >> 5, kk = flat & 31;
            int gm = bm + r, gk = k0 + kk;
            As[r][kk] = (_Float16)((gm < M && gk < K) ? A[(size_t)gm * K + gk] : 0.f);
        }
        // stage B tile 32x64, transposed into Bs[n][k]
        #pragma unroll 4
        for (int i = 0; i < 16; ++i) {
            int flat = tid + i * 128;
            int kk = flat >> 6, n = flat & 63;
            int gk = k0 + kk, gn = bn + n;
            Bs[n][kk] = (_Float16)((gk < K && gn < N) ? B[(size_t)gk * N + gn] : 0.f);
        }
        // prefetch next k-tile into cache while this tile is consumed by WMMA
        if (k0 + 32 < K) {
            int pr = bm + tid;                       // one A row per thread
            if (pr < M) __builtin_prefetch(&A[(size_t)pr * K + k0 + 32], 0, 1);
            int bk = k0 + 32 + (tid & 31);           // one B row per lane
            if (bk < K) __builtin_prefetch(&B[(size_t)bk * N + bn], 0, 1);
        }
        __syncthreads();

        v16h a0, a1;
        {
            int r0 = wave * 32 + ln, r1 = r0 + 16;
            #pragma unroll
            for (int e = 0; e < 16; ++e) {
                int kk = kmap(e, g);
                a0[e] = As[r0][kk];
                a1[e] = As[r1][kk];
            }
        }
        #pragma unroll
        for (int t = 0; t < 4; ++t) {
            v16h b;
            int col = t * 16 + ln;
            #pragma unroll
            for (int e = 0; e < 16; ++e) b[e] = Bs[col][kmap(e, g)];
            acc[0][t] = WMMA_F16(a0, b, acc[0][t]);
            acc[1][t] = WMMA_F16(a1, b, acc[1][t]);
        }
        __syncthreads();
    }

    // epilogue (C/D layout: M = v + 8*g, N = lane%16 within tile)
    #pragma unroll
    for (int half = 0; half < 2; ++half) {
        #pragma unroll
        for (int t = 0; t < 4; ++t) {
            int col = bn + t * 16 + ln;
            if (col >= N) continue;
            float bv = (EP & EP_BIAS) ? bias[col] : 0.f;
            #pragma unroll
            for (int v = 0; v < 8; ++v) {
                int row = bm + wave * 32 + half * 16 + v + 8 * g;
                if (row >= M) continue;
                float x = acc[half][t][v] + bv;
                if (EP & EP_GELU) x = 0.5f * x * (1.f + erff(x * 0.70710678118654752f));
                if (EP & EP_RES)  x += res[(size_t)row * N + col];
                C[(size_t)row * N + col] = x;
            }
        }
    }
}

// ---------------------------------------------------------------------------
// Rowwise LayerNorm: one block per row
// ---------------------------------------------------------------------------
__global__ void __launch_bounds__(256)
layernorm_rows(const float* __restrict__ in, const float* __restrict__ s,
               const float* __restrict__ b, float* __restrict__ out, int C)
{
    __shared__ float ssum[8], ssq[8];
    __shared__ float sm, sinv;
    const int row = blockIdx.x;
    const float* p = in + (size_t)row * C;
    float sum = 0.f, sq = 0.f;
    for (int c = threadIdx.x; c < C; c += blockDim.x) {
        float v = p[c]; sum += v; sq += v * v;
    }
    #pragma unroll
    for (int m = 16; m >= 1; m >>= 1) {
        sum += __shfl_xor(sum, m, 32);
        sq  += __shfl_xor(sq,  m, 32);
    }
    int w = threadIdx.x >> 5;
    if ((threadIdx.x & 31) == 0) { ssum[w] = sum; ssq[w] = sq; }
    __syncthreads();
    if (threadIdx.x == 0) {
        float S = 0.f, Q = 0.f;
        for (int i = 0; i < 8; ++i) { S += ssum[i]; Q += ssq[i]; }
        float mean = S / C;
        float var  = Q / C - mean * mean;
        sm = mean; sinv = rsqrtf(var + 1e-6f);
    }
    __syncthreads();
    float mean = sm, inv = sinv;
    for (int c = threadIdx.x; c < C; c += blockDim.x)
        out[(size_t)row * C + c] = (p[c] - mean) * inv * s[c] + b[c];
}

// ---------------------------------------------------------------------------
// Fused flash-attention with decomposed rel-pos bias, WMMA everywhere.
// grid: (n_windows, 12 heads, q-tile groups of 4); block = 128 (4 waves).
// Each wave owns a 16-query tile; keys processed 32 at a time.
// Windowed layers: winsize=14, L=196, padded (out-of-image) tokens are zero
// features but PRESENT in softmax (matches reference zero-pad-after-LN).
// Global layers: winsize=32, L=1024, one "window".
// ---------------------------------------------------------------------------
__global__ void __launch_bounds__(128)
attention_wmma(const float* __restrict__ qkv, float* __restrict__ out,
               const float* __restrict__ relh, const float* __restrict__ relw,
               int nwx, int winsize, int L)
{
    __shared__ float    sQ[4][16][64];     // unscaled q rows (fp32, feeds bias dots)
    __shared__ float    biasH[4][16][32];
    __shared__ float    biasW[4][16][32];
    __shared__ _Float16 sK[32][64];        // [key][c]
    __shared__ _Float16 sVt[64][32];       // transposed: [c][key]
    __shared__ _Float16 sP[4][16][32];     // per-wave P tile

    const int tid  = threadIdx.x;
    const int wave = tid >> 5;
    const int lane = tid & 31;
    const int g    = lane >> 4;
    const int ln   = lane & 15;
    const int win  = blockIdx.x;
    const int head = blockIdx.y;
    const int wy = win / nwx, wx = win % nwx;
    const int W  = winsize;
    const int n1 = winsize - 1;
    const int qtile0 = blockIdx.z * 4;
    const float scale = 0.125f;            // hd=64 -> 64^-0.5

    auto tok = [&](int p, bool& valid) -> int {
        int i = p / W, j = p % W;
        int gy = wy * winsize + i, gx = wx * winsize + j;
        valid = (gy < 32) && (gx < 32);
        return gy * 32 + gx;
    };

    // ---- stage all 4 q tiles (zero-fill padded / OOR queries) ----
    for (int i = 0; i < 32; ++i) {
        int flat = tid + i * 128;                 // 4096 floats
        int w4 = flat >> 10;
        int r  = (flat >> 6) & 15;
        int c  = flat & 63;
        int p  = (qtile0 + w4) * 16 + r;
        float v = 0.f;
        if (p < L) {
            bool valid; int t = tok(p, valid);
            if (valid) v = qkv[(size_t)t * 2304 + head * 64 + c];
        }
        sQ[w4][r][c] = v;
    }
    __syncthreads();

    // ---- decomposed rel-pos bias: biasH[q][kh]=q.Rh, biasW[q][kw]=q.Rw ----
    int totalBias = 4 * 16 * 2 * winsize;
    for (int u = tid; u < totalBias; u += 128) {
        int w4   = u / (16 * 2 * winsize);
        int rem  = u - w4 * (16 * 2 * winsize);
        int q    = rem / (2 * winsize);
        int rem2 = rem - q * (2 * winsize);
        int isW  = rem2 / winsize;
        int kpos = rem2 % winsize;
        int p    = (qtile0 + w4) * 16 + q;
        float acc = 0.f;
        if (p < L) {
            int qi = p / W, qj = p % W;
            int qq = isW ? qj : qi;
            const float* tab = (isW ? relw : relh) + (size_t)(qq - kpos + n1) * 64;
            const float* qr  = sQ[w4][q];
            for (int c = 0; c < 64; ++c) acc += qr[c] * tab[c];
        }
        if (isW) biasW[w4][q][kpos] = acc; else biasH[w4][q][kpos] = acc;
    }
    __syncthreads();

    // ---- per-wave q in WMMA A-layout, pre-scaled ----
    v16h aQ0, aQ1;
    #pragma unroll
    for (int e = 0; e < 16; ++e) {
        int kk = kmap(e, g);
        aQ0[e] = (_Float16)(sQ[wave][ln][kk]      * scale);
        aQ1[e] = (_Float16)(sQ[wave][ln][kk + 32] * scale);
    }

    float mrow[8], lrow[8];
    #pragma unroll
    for (int v = 0; v < 8; ++v) { mrow[v] = -1e30f; lrow[v] = 0.f; }
    v8f O[4] = {};

    const int nkt = (L + 31) / 32;
    for (int kt = 0; kt < nkt; ++kt) {
        // stage 32 keys of K and V
        for (int i = 0; i < 16; ++i) {
            int flat = tid + i * 128;
            int r = flat >> 6, c = flat & 63;
            int key = kt * 32 + r;
            float kv = 0.f, vv = 0.f;
            if (key < L) {
                bool valid; int t = tok(key, valid);
                if (valid) {
                    kv = qkv[(size_t)t * 2304 + 768  + head * 64 + c];
                    vv = qkv[(size_t)t * 2304 + 1536 + head * 64 + c];
                }
            }
            sK[r][c]  = (_Float16)kv;
            sVt[c][r] = (_Float16)vv;
        }
        // prefetch next K/V tile (one key row per lane -> covers the 256B rows)
        if (kt + 1 < nkt) {
            int key = (kt + 1) * 32 + (tid & 31);
            if (key < L) {
                bool valid; int t = tok(key, valid);
                if (valid) {
                    __builtin_prefetch(qkv + (size_t)t * 2304 + 768  + head * 64, 0, 1);
                    __builtin_prefetch(qkv + (size_t)t * 2304 + 1536 + head * 64, 0, 1);
                }
            }
        }
        __syncthreads();

        // S = q @ k^T for 32 keys (two 16-key accumulators, K=64 split 32+32)
        v8f S0 = {}, S1 = {};
        {
            v16h b;
            #pragma unroll
            for (int e = 0; e < 16; ++e) b[e] = sK[ln][kmap(e, g)];
            S0 = WMMA_F16(aQ0, b, S0);
            #pragma unroll
            for (int e = 0; e < 16; ++e) b[e] = sK[ln][kmap(e, g) + 32];
            S0 = WMMA_F16(aQ1, b, S0);
            #pragma unroll
            for (int e = 0; e < 16; ++e) b[e] = sK[16 + ln][kmap(e, g)];
            S1 = WMMA_F16(aQ0, b, S1);
            #pragma unroll
            for (int e = 0; e < 16; ++e) b[e] = sK[16 + ln][kmap(e, g) + 32];
            S1 = WMMA_F16(aQ1, b, S1);
        }

        // bias + mask + online softmax (row = v + 8*g, key col = lane%16)
        int key0 = kt * 32 + ln, key1 = key0 + 16;
        bool ok0 = key0 < L, ok1 = key1 < L;
        int kh0 = ok0 ? key0 / W : 0, kw0 = ok0 ? key0 % W : 0;
        int kh1 = ok1 ? key1 / W : 0, kw1 = ok1 ? key1 % W : 0;
        #pragma unroll
        for (int v = 0; v < 8; ++v) {
            int M = v + 8 * g;
            float s0 = ok0 ? S0[v] + biasH[wave][M][kh0] + biasW[wave][M][kw0] : -1e30f;
            float s1 = ok1 ? S1[v] + biasH[wave][M][kh1] + biasW[wave][M][kw1] : -1e30f;
            float cm = fmaxf(s0, s1);
            #pragma unroll
            for (int msk = 8; msk >= 1; msk >>= 1) cm = fmaxf(cm, __shfl_xor(cm, msk, 32));
            float mn    = fmaxf(mrow[v], cm);
            float alpha = __expf(mrow[v] - mn);
            float p0 = ok0 ? __expf(s0 - mn) : 0.f;
            float p1 = ok1 ? __expf(s1 - mn) : 0.f;
            float ps = p0 + p1;
            #pragma unroll
            for (int msk = 8; msk >= 1; msk >>= 1) ps += __shfl_xor(ps, msk, 32);
            lrow[v] = lrow[v] * alpha + ps;
            mrow[v] = mn;
            #pragma unroll
            for (int t = 0; t < 4; ++t) O[t][v] *= alpha;
            sP[wave][M][ln]      = (_Float16)p0;
            sP[wave][M][ln + 16] = (_Float16)p1;
        }

        // O += P @ V   (P: 16x32 A-layout; V: 32x64 as 4 B-tiles)
        v16h aP;
        #pragma unroll
        for (int e = 0; e < 16; ++e) aP[e] = sP[wave][ln][kmap(e, g)];
        #pragma unroll
        for (int t = 0; t < 4; ++t) {
            v16h b;
            #pragma unroll
            for (int e = 0; e < 16; ++e) b[e] = sVt[t * 16 + ln][kmap(e, g)];
            O[t] = WMMA_F16(aP, b, O[t]);
        }
        __syncthreads();
    }

    // normalize + scatter (window-unpartition inline; only in-image tokens)
    #pragma unroll
    for (int v = 0; v < 8; ++v) {
        int M = v + 8 * g;
        int p = (qtile0 + wave) * 16 + M;
        if (p >= L) continue;
        bool valid; int t_ = tok(p, valid);
        if (!valid) continue;
        float inv = 1.f / lrow[v];
        #pragma unroll
        for (int t = 0; t < 4; ++t)
            out[(size_t)t_ * 768 + head * 64 + t * 16 + ln] = O[t][v] * inv;
    }
}

// ---------------------------------------------------------------------------
// small data-movement kernels
// ---------------------------------------------------------------------------
__global__ void patch_gather(const float* __restrict__ x, float* __restrict__ A)
{
    int idx = blockIdx.x * blockDim.x + threadIdx.x;
    if (idx >= 1024 * 768) return;
    int t = idx / 768, k = idx % 768;
    int py = t >> 5, px = t & 31;
    int c = k % 3, pp = k / 3;
    int pw = pp & 15, ph = pp >> 4;
    A[idx] = x[c * 512 * 512 + (py * 16 + ph) * 512 + (px * 16 + pw)];
}

__global__ void neck_im2col(const float* __restrict__ in, float* __restrict__ col)
{
    int idx = blockIdx.x * blockDim.x + threadIdx.x;
    if (idx >= 1024 * 2304) return;
    int t = idx / 2304, k = idx % 2304;
    int y = t >> 5, x = t & 31;
    int ci = k % 256, kk = k / 256;
    int ky = kk / 3, kx = kk % 3;
    int yy = y + ky - 1, xx = x + kx - 1;
    float v = 0.f;
    if (yy >= 0 && yy < 32 && xx >= 0 && xx < 32) v = in[(size_t)((yy << 5) + xx) * 256 + ci];
    col[idx] = v;
}

__global__ void to_nchw(const float* __restrict__ in, float* __restrict__ out)
{
    int idx = blockIdx.x * blockDim.x + threadIdx.x;
    if (idx >= 256 * 1024) return;
    int c = idx / 1024, t = idx % 1024;
    out[idx] = in[(size_t)t * 256 + c];
}

// ---------------------------------------------------------------------------
// host launcher
// ---------------------------------------------------------------------------
extern "C" void kernel_launch(void* const* d_in, const int* in_sizes, int n_in,
                              void* d_out, int out_size, void* d_ws, size_t ws_size,
                              hipStream_t stream)
{
    (void)in_sizes; (void)n_in; (void)out_size; (void)ws_size;
    const float* x          = (const float*)d_in[0];
    const float* patch_w    = (const float*)d_in[1];
    const float* patch_b    = (const float*)d_in[2];
    const float* pos_embed  = (const float*)d_in[3];
    const float* ln1_s      = (const float*)d_in[4];
    const float* ln1_b      = (const float*)d_in[5];
    const float* qkv_w      = (const float*)d_in[6];
    const float* qkv_b      = (const float*)d_in[7];
    const float* proj_w     = (const float*)d_in[8];
    const float* proj_b     = (const float*)d_in[9];
    const float* relh_win   = (const float*)d_in[10];
    const float* relw_win   = (const float*)d_in[11];
    const float* relh_glob  = (const float*)d_in[12];
    const float* relw_glob  = (const float*)d_in[13];
    const float* ln2_s      = (const float*)d_in[14];
    const float* ln2_b      = (const float*)d_in[15];
    const float* fc1_w      = (const float*)d_in[16];
    const float* fc1_b      = (const float*)d_in[17];
    const float* fc2_w      = (const float*)d_in[18];
    const float* fc2_b      = (const float*)d_in[19];
    const float* neck_w1    = (const float*)d_in[20];
    const float* neck_ln1_s = (const float*)d_in[21];
    const float* neck_ln1_b = (const float*)d_in[22];
    const float* neck_w2    = (const float*)d_in[23];
    const float* neck_ln2_s = (const float*)d_in[24];
    const float* neck_ln2_b = (const float*)d_in[25];

    float* ws    = (float*)d_ws;
    float* h     = ws; ws += 1024 * 768;
    float* y     = ws; ws += 1024 * 768;
    float* qkv   = ws; ws += 1024 * 2304;
    float* attno = ws; ws += 1024 * 768;
    float* mlp1  = ws; ws += 1024 * 3072;
    float* colb  = ws; ws += 1024 * 2304;
    float* na    = ws; ws += 1024 * 256;
    float* nb    = ws; ws += 1024 * 256;

    // patch embed: im2col + GEMM (+patch_b bias, +pos_embed as residual)
    patch_gather<<<(1024 * 768 + 255) / 256, 256, 0, stream>>>(x, colb);
    gemm_f16wmma<EP_BIAS | EP_RES><<<dim3(12, 8), 128, 0, stream>>>(
        colb, patch_w, patch_b, pos_embed, h, 1024, 768, 768);

    int wi = 0, gi = 0;
    for (int i = 0; i < 12; ++i) {
        bool isg = (i == 2 || i == 5 || i == 8 || i == 11);
        layernorm_rows<<<1024, 256, 0, stream>>>(h, ln1_s + i * 768, ln1_b + i * 768, y, 768);
        gemm_f16wmma<EP_BIAS><<<dim3(36, 8), 128, 0, stream>>>(
            y, qkv_w + (size_t)i * 768 * 2304, qkv_b + i * 2304, nullptr, qkv, 1024, 2304, 768);
        if (isg) {
            attention_wmma<<<dim3(1, 12, 16), 128, 0, stream>>>(
                qkv, attno, relh_glob + (size_t)gi * 63 * 64, relw_glob + (size_t)gi * 63 * 64,
                1, 32, 1024);
            ++gi;
        } else {
            attention_wmma<<<dim3(9, 12, 4), 128, 0, stream>>>(
                qkv, attno, relh_win + (size_t)wi * 27 * 64, relw_win + (size_t)wi * 27 * 64,
                3, 14, 196);
            ++wi;
        }
        gemm_f16wmma<EP_BIAS | EP_RES><<<dim3(12, 8), 128, 0, stream>>>(
            attno, proj_w + (size_t)i * 768 * 768, proj_b + i * 768, h, h, 1024, 768, 768);
        layernorm_rows<<<1024, 256, 0, stream>>>(h, ln2_s + i * 768, ln2_b + i * 768, y, 768);
        gemm_f16wmma<EP_BIAS | EP_GELU><<<dim3(48, 8), 128, 0, stream>>>(
            y, fc1_w + (size_t)i * 768 * 3072, fc1_b + i * 3072, nullptr, mlp1, 1024, 3072, 768);
        gemm_f16wmma<EP_BIAS | EP_RES><<<dim3(12, 8), 128, 0, stream>>>(
            mlp1, fc2_w + (size_t)i * 3072 * 768, fc2_b + i * 768, h, h, 1024, 768, 3072);
    }

    // neck: 1x1 conv -> LN -> 3x3 conv -> LN -> NCHW
    gemm_f16wmma<0><<<dim3(4, 8), 128, 0, stream>>>(h, neck_w1, nullptr, nullptr, na, 1024, 256, 768);
    layernorm_rows<<<1024, 256, 0, stream>>>(na, neck_ln1_s, neck_ln1_b, nb, 256);
    neck_im2col<<<(1024 * 2304 + 255) / 256, 256, 0, stream>>>(nb, colb);
    gemm_f16wmma<0><<<dim3(4, 8), 128, 0, stream>>>(colb, neck_w2, nullptr, nullptr, na, 1024, 256, 2304);
    layernorm_rows<<<1024, 256, 0, stream>>>(na, neck_ln2_s, neck_ln2_b, nb, 256);
    to_nchw<<<(256 * 1024 + 255) / 256, 256, 0, stream>>>(nb, (float*)d_out);
}
